// MultiHeadAttention_78632261255411
// MI455X (gfx1250) — matrix operand order repaired and checked by reading, not compiled, not run on hardware
//
#include <hip/hip_runtime.h>
#include <math.h>

typedef __bf16 bf16;
typedef __bf16 bf16x8 __attribute__((ext_vector_type(8)));
typedef __bf16 v16bf  __attribute__((ext_vector_type(16)));
typedef float  v8f    __attribute__((ext_vector_type(8)));

static constexpr int Bb = 2, S = 2048, Dm = 1024, E = 1024, H = 16, HD = 64;
static constexpr int M_TOK  = Bb * S;    // 4096 token rows
static constexpr int THREEE = 3 * E;     // 3072

__device__ __forceinline__ v16bf cat16(bf16x8 lo, bf16x8 hi) {
  return __builtin_shufflevector(lo, hi, 0,1,2,3,4,5,6,7,8,9,10,11,12,13,14,15);
}

__device__ __forceinline__ v8f wmma_bf16(v16bf a, v16bf b, v8f c) {
  return __builtin_amdgcn_wmma_f32_16x16x32_bf16(false, a, false, b, (short)0, c,
                                                 false, false);
}

// ---------------------------------------------------------------- convert
__global__ __launch_bounds__(256) void f32_to_bf16_kernel(
    const float* __restrict__ src, bf16* __restrict__ dst, int n) {
  int i = blockIdx.x * blockDim.x + threadIdx.x;
  int stride = gridDim.x * blockDim.x;
  for (; i < n; i += stride) dst[i] = (bf16)src[i];
}

// ---------------------------------------------------------------- GEMM core
// 32x64 tile per wave at (m0, n0): A[M x K] row-major, W[N x K] row-major
// (row n of W = weights of output column n).  All fragments of a K-chunk are
// loaded before the 8 WMMAs so VMEM latency overlaps matrix ops.
__device__ __forceinline__ void gemm_tile_32x64(
    const bf16* __restrict__ A, const bf16* __restrict__ W,
    int m0, int n0, int K, v8f acc[2][4]) {
  const int lane = threadIdx.x & 31;
  const int col  = lane & 15;
  const int hi   = lane >> 4;
  const bf16* arow0 = A + (size_t)(m0 + col) * K;
  const bf16* arow1 = A + (size_t)(m0 + 16 + col) * K;
  for (int k = 0; k < K; k += 32) {
    __builtin_prefetch(arow0 + k + 512, 0, 1);  // speculative; OOB dropped
    bf16x8 a00 = *(const bf16x8*)(arow0 + k + hi * 8);
    bf16x8 a01 = *(const bf16x8*)(arow0 + k + 16 + hi * 8);
    bf16x8 a10 = *(const bf16x8*)(arow1 + k + hi * 8);
    bf16x8 a11 = *(const bf16x8*)(arow1 + k + 16 + hi * 8);
    v16bf af0 = cat16(a00, a01);
    v16bf af1 = cat16(a10, a11);
    v16bf bfr[4];
#pragma unroll
    for (int t = 0; t < 4; ++t) {
      const bf16* wrow = W + (size_t)(n0 + t * 16 + col) * K + k + hi * 8;
      bfr[t] = cat16(*(const bf16x8*)(wrow), *(const bf16x8*)(wrow + 16));
    }
#pragma unroll
    for (int t = 0; t < 4; ++t) {
      acc[0][t] = wmma_bf16(af0, bfr[t], acc[0][t]);
      acc[1][t] = wmma_bf16(af1, bfr[t], acc[1][t]);
    }
  }
}

// QKV projection: C bf16 [M x N], + bias.  Block = 8 waves = 64M x 256N.
__global__ __launch_bounds__(256) void gemm_bf16out_kernel(
    const bf16* __restrict__ A, const bf16* __restrict__ W,
    const float* __restrict__ bias, bf16* __restrict__ C, int N, int K) {
  const int wave = threadIdx.x >> 5, lane = threadIdx.x & 31;
  const int m0 = blockIdx.y * 64 + (wave & 1) * 32;
  const int n0 = blockIdx.x * 256 + (wave >> 1) * 64;
  v8f acc[2][4] = {};
  gemm_tile_32x64(A, W, m0, n0, K, acc);
  const int col = lane & 15, hi = lane >> 4;
#pragma unroll
  for (int s2 = 0; s2 < 2; ++s2) {
#pragma unroll
    for (int t = 0; t < 4; ++t) {
      int n = n0 + t * 16 + col;
      float bv = bias[n];
#pragma unroll
      for (int r = 0; r < 8; ++r) {
        int m = m0 + s2 * 16 + r + hi * 8;
        C[(size_t)m * N + n] = (bf16)(acc[s2][t][r] + bv);
      }
    }
  }
}

// Output projection: C fp32 [M x N], + bias.
__global__ __launch_bounds__(256) void gemm_f32out_kernel(
    const bf16* __restrict__ A, const bf16* __restrict__ W,
    const float* __restrict__ bias, float* __restrict__ C, int N, int K) {
  const int wave = threadIdx.x >> 5, lane = threadIdx.x & 31;
  const int m0 = blockIdx.y * 64 + (wave & 1) * 32;
  const int n0 = blockIdx.x * 256 + (wave >> 1) * 64;
  v8f acc[2][4] = {};
  gemm_tile_32x64(A, W, m0, n0, K, acc);
  const int col = lane & 15, hi = lane >> 4;
#pragma unroll
  for (int s2 = 0; s2 < 2; ++s2) {
#pragma unroll
    for (int t = 0; t < 4; ++t) {
      int n = n0 + t * 16 + col;
      float bv = bias[n];
#pragma unroll
      for (int r = 0; r < 8; ++r) {
        int m = m0 + s2 * 16 + r + hi * 8;
        C[(size_t)m * N + n] = acc[s2][t][r] + bv;
      }
    }
  }
}

// ---------------------------------------------------------------- attention
// QKV layout per token row (3072 cols): head h -> [q(64) | k(64) | v(64)] at
// col h*192 (matches the reference's reshape(b,s,H,3*HD) semantics).
// One block = one (b,h) and 128 query rows (8 waves x 16 rows each).
__global__ __launch_bounds__(256) void attention_kernel(
    const bf16* __restrict__ QKV, float* __restrict__ attn,
    bf16* __restrict__ vals) {
  __shared__ bf16 Vt[HD * 32];        // V chunk transposed: [d][key] 64x32
  __shared__ bf16 Pbuf[8][16 * 32];   // per-wave P staging [m][key]

  const int wave = threadIdx.x >> 5, lane = threadIdx.x & 31;
  const int col = lane & 15, hi = lane >> 4;
  const int qchunk = blockIdx.x & 15;       // S/128 = 16 chunks
  const int bh = blockIdx.x >> 4;           // b*16 + h
  const int b = bh >> 4, h = bh & 15;
  const int q0 = qchunk * 128 + wave * 16;

  const bf16* Qbase = QKV + (size_t)(b * S) * THREEE + h * (3 * HD);
  const bf16* Kbase = Qbase + HD;
  const bf16* Vbase = Qbase + 2 * HD;

  // Q A-fragments (two 32-wide d chunks); A row m = col
  const bf16* qrow = Qbase + (size_t)(q0 + col) * THREEE;
  v16bf aq[2];
#pragma unroll
  for (int c = 0; c < 2; ++c) {
    bf16x8 a0 = *(const bf16x8*)(qrow + c * 32 + hi * 8);
    bf16x8 a1 = *(const bf16x8*)(qrow + c * 32 + 16 + hi * 8);
    aq[c] = cat16(a0, a1);
  }

  const float inv_hd = 1.0f / 64.0f;

  // logits tile for 16 keys at key0: load both K-chunk B-frags, then 2 WMMAs
  auto logits = [&](int key0) -> v8f {
    const bf16* krow = Kbase + (size_t)(key0 + col) * THREEE + hi * 8;
    v16bf bk0 = cat16(*(const bf16x8*)(krow), *(const bf16x8*)(krow + 16));
    v16bf bk1 = cat16(*(const bf16x8*)(krow + 32), *(const bf16x8*)(krow + 48));
    v8f c = {};
    c = wmma_bf16(aq[0], bk0, c);
    c = wmma_bf16(aq[1], bk1, c);
    return c;
  };

  // ---- pass 1: online row max / sum-of-exp (logits discarded) ----
  float rmax[8], rsum[8];
#pragma unroll
  for (int r = 0; r < 8; ++r) { rmax[r] = -INFINITY; rsum[r] = 0.0f; }

  for (int key0 = 0; key0 < S; key0 += 16) {
    v8f c = logits(key0);
#pragma unroll
    for (int r = 0; r < 8; ++r) {
      float x = c[r] * inv_hd;
      float tm = x;
#pragma unroll
      for (int off = 1; off < 16; off <<= 1)
        tm = fmaxf(tm, __shfl_xor(tm, off, 16));
      float mnew = fmaxf(rmax[r], tm);
      float e = __expf(x - mnew);
#pragma unroll
      for (int off = 1; off < 16; off <<= 1) e += __shfl_xor(e, off, 16);
      rsum[r] = rsum[r] * __expf(rmax[r] - mnew) + e;
      rmax[r] = mnew;
    }
  }
  float rinv[8];
#pragma unroll
  for (int r = 0; r < 8; ++r) rinv[r] = 1.0f / rsum[r];

  // ---- pass 2: recompute logits, write attn once (NT), attn @ V ----
  v8f acc[4] = {};
  bf16* pb = &Pbuf[wave][0];

  for (int kc = 0; kc < S; kc += 32) {
    // cooperative stage of V chunk, transposed into LDS
    __syncthreads();
    {
      int t = threadIdx.x;
      int key = t >> 3;                 // 0..31
      int ds = (t & 7) * 8;             // 0..56
      bf16x8 vv =
          *(const bf16x8*)(Vbase + (size_t)(kc + key) * THREEE + ds);
#pragma unroll
      for (int j = 0; j < 8; ++j) Vt[(ds + j) * 32 + key] = vv[j];
    }
    __syncthreads();

#pragma unroll
    for (int jj = 0; jj < 2; ++jj) {
      int key0 = kc + jj * 16;
      v8f c = logits(key0);
#pragma unroll
      for (int r = 0; r < 8; ++r) {
        int m = r + hi * 8;
        float p = __expf(c[r] * inv_hd - rmax[r]) * rinv[r];
        __builtin_nontemporal_store(
            p, attn + ((size_t)bh * S + q0 + m) * S + key0 + col);
        pb[m * 32 + jj * 16 + col] = (bf16)p;   // C-layout -> [m][key] in LDS
      }
    }

    // A-fragment of P from LDS (row m = col)
    bf16x8 p0 = *(const bf16x8*)(pb + col * 32 + hi * 8);
    bf16x8 p1 = *(const bf16x8*)(pb + col * 32 + 16 + hi * 8);
    v16bf pa = cat16(p0, p1);

    // B-fragments of V from transposed LDS (all 4 first), then 4 WMMAs
    v16bf bv[4];
#pragma unroll
    for (int t4 = 0; t4 < 4; ++t4) {
      const bf16* vtrow = Vt + (t4 * 16 + col) * 32 + hi * 8;
      bv[t4] = cat16(*(const bf16x8*)(vtrow), *(const bf16x8*)(vtrow + 16));
    }
#pragma unroll
    for (int t4 = 0; t4 < 4; ++t4) acc[t4] = wmma_bf16(pa, bv[t4], acc[t4]);
  }

  // store vals (bf16) in [b, s, h*64 + d] layout for the output GEMM
#pragma unroll
  for (int t4 = 0; t4 < 4; ++t4) {
#pragma unroll
    for (int r = 0; r < 8; ++r) {
      int m = q0 + r + hi * 8;
      int e = h * 64 + t4 * 16 + col;
      vals[(size_t)(b * S + m) * E + e] = (bf16)acc[t4][r];
    }
  }
}

// ---------------------------------------------------------------- launch
extern "C" void kernel_launch(void* const* d_in, const int* in_sizes, int n_in,
                              void* d_out, int out_size, void* d_ws,
                              size_t ws_size, hipStream_t stream) {
  const float* x     = (const float*)d_in[0];  // [4096,1024]
  const float* w_qkv = (const float*)d_in[1];  // [3072,1024]
  const float* b_qkv = (const float*)d_in[2];  // [3072]
  const float* w_out = (const float*)d_in[3];  // [1024,1024]
  const float* b_out = (const float*)d_in[4];  // [1024]

  float* out  = (float*)d_out;                         // [4096,1024]
  float* attn = out + (size_t)M_TOK * Dm;              // [32,2048,2048]

  bf16* xh   = (bf16*)d_ws;                            //  8 MB
  bf16* wqh  = xh  + (size_t)M_TOK * Dm;               //  6 MB
  bf16* woh  = wqh + (size_t)THREEE * Dm;              //  2 MB
  bf16* qkv  = woh + (size_t)Dm * E;                   // 24 MB
  bf16* vals = qkv + (size_t)M_TOK * THREEE;           //  8 MB

  // fp32 -> bf16 conversions
  f32_to_bf16_kernel<<<2048, 256, 0, stream>>>(x, xh, M_TOK * Dm);
  f32_to_bf16_kernel<<<2048, 256, 0, stream>>>(w_qkv, wqh, THREEE * Dm);
  f32_to_bf16_kernel<<<1024, 256, 0, stream>>>(w_out, woh, Dm * E);

  // QKV projection: [4096,3072] = xh @ wqh^T + b_qkv
  gemm_bf16out_kernel<<<dim3(THREEE / 256, M_TOK / 64), 256, 0, stream>>>(
      xh, wqh, b_qkv, qkv, THREEE, Dm);

  // attention: softmax(Q K^T / 64) -> attn (written once), vals = attn @ V
  attention_kernel<<<Bb * H * (S / 128), 256, 0, stream>>>(qkv, attn, vals);

  // output projection: [4096,1024] = vals @ woh^T + b_out
  gemm_f32out_kernel<<<dim3(Dm / 256, M_TOK / 64), 256, 0, stream>>>(
      vals, woh, b_out, out, Dm, E);
}
